// MultiheadAttention_50637664420366
// MI455X (gfx1250) — matrix operand order repaired and checked by reading, not compiled, not run on hardware
//
#include <hip/hip_runtime.h>

// ---------------------------------------------------------------------------
// Transformer block for MI455X (gfx1250): all matmuls via v_wmma_f32_16x16x32_bf16,
// GEMM weight tiles staged to LDS by the Tensor Data Mover (tensor_load_to_lds,
// TENSORcnt-synchronized, double-buffered).
// B=16, C=d=512, H=W=32 -> S=1024 tokens, T=16384 total tokens, 8 heads x 64
// ---------------------------------------------------------------------------

typedef __bf16 bf16_t;
typedef __attribute__((ext_vector_type(16))) __bf16 v16bf;
typedef __attribute__((ext_vector_type(8)))  __bf16 v8bf;
typedef __attribute__((ext_vector_type(8)))  float  v8f;
typedef __attribute__((ext_vector_type(4)))  unsigned int u32x4;
typedef __attribute__((ext_vector_type(4)))  int i32x4;
typedef __attribute__((ext_vector_type(8)))  int i32x8;

#define D_MODEL 512
#define SEQ     1024
#define BATCH   16
#define TOK     (BATCH * SEQ)   // 16384
#define NHEAD   8
#define HDIM    64

#if __has_builtin(__builtin_amdgcn_tensor_load_to_lds)
#define HAVE_TDM 1
#else
#define HAVE_TDM 0
#endif

__device__ __forceinline__ v8f wmma_bf16(v16bf a, v16bf b, v8f c) {
  // D = A(16x32 bf16) * B(32x16 bf16) + C(16x16 f32)
  return __builtin_amdgcn_wmma_f32_16x16x32_bf16(
      /*neg_a=*/false, a, /*neg_b=*/false, b,
      /*c_mod=*/(short)0, c, /*reuse_a=*/false, /*reuse_b=*/false);
}

#if HAVE_TDM
// TDM: DMA a 32-row x 256-col bf16 tile of a 512-col matrix into LDS.
// D# group0: count=1 | lds_addr[63:32] | global_addr[120:64] | type=2[127:126]
// D# group1: data_size=1(2B)[17:16], tensor_dim0=512[79:48], tensor_dim1=512[111:80],
//            tile_dim0=256[127:112], tile_dim1=32[143:128], dim0_stride=512[207:160]
__device__ __forceinline__ void tdm_load_w_tile(const bf16_t* gptr, unsigned lds_off) {
  unsigned long long ga = (unsigned long long)(size_t)gptr;
  u32x4 g0;
  g0[0] = 1u;                                            // count=1 (valid user D#)
  g0[1] = lds_off;                                       // lds_addr
  g0[2] = (unsigned)(ga & 0xFFFFFFFFu);                  // global_addr[31:0]
  g0[3] = (unsigned)((ga >> 32) & 0x1FFFFFFu) | (2u << 30);  // addr[56:32] | type=2
  i32x8 g1;
  g1[0] = 1 << 16;                                       // data_size = 1 -> 2 bytes
  g1[1] = (int)(512u << 16);                             // tensor_dim0 = 512
  g1[2] = (int)(512u << 16);                             // tensor_dim1 = 512
  g1[3] = (int)(256u << 16);                             // tile_dim0   = 256
  g1[4] = 32;                                            // tile_dim1   = 32
  g1[5] = 512;                                           // tensor_dim0_stride = 512
  g1[6] = 0;
  g1[7] = 0;
  i32x4 z4 = {0, 0, 0, 0};
#if defined(__clang_major__) && __clang_major__ >= 23
  i32x8 z8 = {0, 0, 0, 0, 0, 0, 0, 0};
  __builtin_amdgcn_tensor_load_to_lds(g0, g1, z4, z4, z8, 0);
#else
  __builtin_amdgcn_tensor_load_to_lds(g0, g1, z4, z4, 0);
#endif
}
#endif

// ---------------------------------------------------------------------------
// Weight repack: fp32 -> bf16, K-major [512 x 512]
// ---------------------------------------------------------------------------
__global__ __launch_bounds__(256) void prep_weights(
    const float* Qw, const float* Kw, const float* Vw,
    const float* Ow, const float* W1, const float* W2,
    bf16_t* wq, bf16_t* wk, bf16_t* wv, bf16_t* wo, bf16_t* w1t, bf16_t* w2t)
{
  int e = blockIdx.x * 256 + threadIdx.x;     // 0 .. 512*512-1
  int c = e >> 9;
  int n = e & 511;
  int qkv_idx = (n >> 6) * (D_MODEL * HDIM) + c * HDIM + (n & 63);
  switch (blockIdx.y) {
    case 0: wq[e]  = (bf16_t)Qw[qkv_idx];     break;
    case 1: wk[e]  = (bf16_t)Kw[qkv_idx];     break;
    case 2: wv[e]  = (bf16_t)Vw[qkv_idx];     break;
    case 3: wo[e]  = (bf16_t)Ow[e];           break;
    case 4: w1t[e] = (bf16_t)W1[n * 512 + c]; break;
    default: w2t[e] = (bf16_t)W2[n * 512 + c]; break;
  }
}

// ---------------------------------------------------------------------------
// LayerNorm over C=512, one wave32 per token; shfl_xor tree reductions.
// ---------------------------------------------------------------------------
template<bool STRIDED>
__global__ __launch_bounds__(256) void ln_kernel(
    const float* __restrict__ in, const float* __restrict__ gamma,
    const float* __restrict__ beta, bf16_t* __restrict__ out)
{
  int wave = (blockIdx.x * 256 + threadIdx.x) >> 5;   // token id
  int lane = threadIdx.x & 31;
  int bb = wave >> 10, s = wave & 1023;

  float vals[16];
  float sum = 0.f;
#pragma unroll
  for (int i = 0; i < 16; ++i) {
    int c = i * 32 + lane;
    float xv = STRIDED ? in[(size_t)bb * (D_MODEL * SEQ) + c * SEQ + s]
                       : in[(size_t)wave * D_MODEL + c];
    vals[i] = xv;
    sum += xv;
  }
#pragma unroll
  for (int off = 16; off > 0; off >>= 1) sum += __shfl_xor(sum, off, 32);
  float mu = sum * (1.0f / 512.0f);

  float var = 0.f;
#pragma unroll
  for (int i = 0; i < 16; ++i) { float d = vals[i] - mu; var += d * d; }
#pragma unroll
  for (int off = 16; off > 0; off >>= 1) var += __shfl_xor(var, off, 32);
  float r = rsqrtf(var * (1.0f / 512.0f) + 1e-5f);

#pragma unroll
  for (int i = 0; i < 16; ++i) {
    int c = i * 32 + lane;
    out[(size_t)wave * D_MODEL + c] = (bf16_t)((vals[i] - mu) * r * gamma[c] + beta[c]);
  }
}

// ---------------------------------------------------------------------------
// bf16 WMMA GEMM: out[16384 x 512] = A[16384 x 512] * W[512 x 512]
// Block = 4 waves sharing one 16-row A tile and a 256-col W slab.
// W k-slabs (32x256 bf16 = 16 KB) are DMA'd to LDS by the TDM, double-buffered:
//   leader wave issues tensor_load_to_lds for tile k+1 right after the barrier
//   that publishes tile k, so the DMA overlaps this iteration's 16 WMMAs.
// Epilogues: 0 = +bias -> bf16 (QKV)
//            1 = +bias + x[B,C,S] residual -> f32 (O-proj)
//            2 = +bias + exact GELU -> bf16 (MLP fc1)
//            3 = +bias + f32 residual -> f32 stored transposed to [B,C,S]
// ---------------------------------------------------------------------------
template<int EPI>
__global__ __launch_bounds__(128) void gemm_wmma(
    const bf16_t* __restrict__ A, const bf16_t* __restrict__ W,
    const float* __restrict__ bias, const float* __restrict__ resid,
    bf16_t* __restrict__ out_bf, float* __restrict__ out_f)
{
  __shared__ __align__(32) bf16_t sW[2][32][256];   // double-buffered W slab

  int lane = threadIdx.x & 31;
  int ntl  = threadIdx.x >> 5;                      // wave-local n-tile 0..3
  int mt    = blockIdx.x >> 1;                      // 16-row tile 0..1023
  int nbase = (blockIdx.x & 1) * 256;               // 256-col slab
  int rlane = lane & 15;
  int hsel  = lane >> 4;

  const bf16_t* wbase = W + nbase;                  // (row 0, col nbase)
  bool leader = (threadIdx.x < 32);                 // wave 0 drives the TDM

  int arow = mt * 16 + rlane;
  const bf16_t* arp = A + (size_t)arow * D_MODEL;

#if HAVE_TDM
  unsigned lds_buf[2] = { (unsigned)(size_t)&sW[0][0][0],
                          (unsigned)(size_t)&sW[1][0][0] };
  if (leader) tdm_load_w_tile(wbase, lds_buf[0]);   // prologue: tile k0=0
#else
  {  // fallback: cooperative copy of tile k0=0
    for (int e = threadIdx.x * 8; e < 32 * 256; e += 128 * 8)
      *(v8bf*)&sW[0][0][e] = *(const v8bf*)(wbase + (size_t)(e >> 8) * D_MODEL + (e & 255));
  }
#endif

  v8f acc[4] = { {}, {}, {}, {} };
  int cur = 0;

  for (int k0 = 0; k0 < D_MODEL; k0 += 32) {
#if HAVE_TDM
    if (leader) __builtin_amdgcn_s_wait_tensorcnt(0);   // buf[cur] DMA complete
#endif
    __syncthreads();                                    // publish buf[cur]; retire reads of buf[cur^1]
#if HAVE_TDM
    if (leader && (k0 + 32 < D_MODEL))
      tdm_load_w_tile(wbase + (size_t)(k0 + 32) * D_MODEL, lds_buf[cur ^ 1]);
#else
    if (k0 + 32 < D_MODEL)
      for (int e = threadIdx.x * 8; e < 32 * 256; e += 128 * 8)
        *(v8bf*)&sW[cur ^ 1][0][e] =
            *(const v8bf*)(wbase + (size_t)(k0 + 32 + (e >> 8)) * D_MODEL + (e & 255));
#endif

    // A-frag from global: lane<16 holds K=k0+{0..7,16..23}; lane>=16 +8
    int ka = k0 + hsel * 8;
    v8bf alo = *(const v8bf*)(arp + ka);
    v8bf ahi = *(const v8bf*)(arp + ka + 16);
    if (k0 + 32 < D_MODEL) __builtin_prefetch(arp + k0 + 32, 0, 1);
    v16bf afrag;
#pragma unroll
    for (int i = 0; i < 8; ++i) { afrag[i] = alo[i]; afrag[8 + i] = ahi[i]; }

    // B-frags from LDS: lane -> col rlane; element s -> K-row k0+{0..7,16..23}(+8 for lane>=16)
    const bf16_t* wl = &sW[cur][0][ntl * 64 + rlane];
#pragma unroll
    for (int nb = 0; nb < 4; ++nb) {
      v16bf bfrag;
#pragma unroll
      for (int s = 0; s < 16; ++s)
        bfrag[s] = wl[(s < 8 ? hsel * 8 + s : 16 + hsel * 8 + (s - 8)) * 256 + nb * 16];
      acc[nb] = wmma_bf16(afrag, bfrag, acc[nb]);
    }
    cur ^= 1;
  }

  // Epilogue: C-frag row = mt*16 + r + hsel*8 ; col = nbase + ntl*64 + nb*16 + rlane
  int rbase = mt * 16 + hsel * 8;
#pragma unroll
  for (int nb = 0; nb < 4; ++nb) {
    int nc = nbase + ntl * 64 + nb * 16 + rlane;
    float bs = bias[nc];
#pragma unroll
    for (int r = 0; r < 8; ++r) {
      int rr = rbase + r;                       // token id
      float v = acc[nb][r] + bs;
      if (EPI == 0) {
        out_bf[(size_t)rr * D_MODEL + nc] = (bf16_t)v;
      } else if (EPI == 1) {
        int b = rr >> 10, s = rr & 1023;
        out_f[(size_t)rr * D_MODEL + nc] =
            v + resid[(size_t)b * (D_MODEL * SEQ) + nc * SEQ + s];
      } else if (EPI == 2) {
        float g = 0.5f * v * (1.0f + erff(v * 0.70710678118654752f));
        out_bf[(size_t)rr * D_MODEL + nc] = (bf16_t)g;
      } else {
        int b = rr >> 10, s = rr & 1023;
        out_f[(size_t)b * (D_MODEL * SEQ) + nc * SEQ + s] =
            v + resid[(size_t)rr * D_MODEL + nc];
      }
    }
  }
}

// ---------------------------------------------------------------------------
// Flash attention, one wave per (b, h, 16 query rows).
// Scores computed TRANSPOSED: T = K_tile(16x64) * Q^T(64x16) so the f32
// C-layout of T maps element-for-element onto the bf16 A-frag layout of P for
// the P*V WMMA (no cross-lane movement in the softmax->A repack).
// ---------------------------------------------------------------------------
__global__ __launch_bounds__(128) void attn_kernel(
    const bf16_t* __restrict__ q, const bf16_t* __restrict__ k,
    const bf16_t* __restrict__ v, bf16_t* __restrict__ o)
{
  __shared__ __align__(32) bf16_t sQ[4][16 * HDIM];

  int lwave = threadIdx.x >> 5;
  int lane  = threadIdx.x & 31;
  int wave  = blockIdx.x * 4 + lwave;
  int b  = wave >> 9;           // / (8*64)
  int h  = (wave >> 6) & 7;
  int it = wave & 63;           // 16-row query block
  int qrow0 = it * 16;
  int rlane = lane & 15;
  int hsel  = lane >> 4;

  const bf16_t* qb = q + ((size_t)b * SEQ) * D_MODEL + h * HDIM;
  const bf16_t* kb = k + ((size_t)b * SEQ) * D_MODEL + h * HDIM;
  const bf16_t* vb = v + ((size_t)b * SEQ) * D_MODEL + h * HDIM;

  // Stage the 16x64 Q tile into LDS (coalesced), then gather Q^T B-frags.
  {
    const bf16_t* src = qb + (size_t)(qrow0 + rlane) * D_MODEL + hsel * 32;
    v16bf t0 = *(const v16bf*)(src);
    v16bf t1 = *(const v16bf*)(src + 16);
    *(v16bf*)&sQ[lwave][rlane * HDIM + hsel * 32]      = t0;
    *(v16bf*)&sQ[lwave][rlane * HDIM + hsel * 32 + 16] = t1;
  }
  // B = Q^T (64x16): two K=32 frags; lane -> q-row rlane, element e -> d = dc*32 + table-order k
  v16bf qt[2];
#pragma unroll
  for (int dc = 0; dc < 2; ++dc)
#pragma unroll
    for (int e = 0; e < 16; ++e)
      qt[dc][e] = sQ[lwave][rlane * HDIM + dc * 32 + (e < 8 ? hsel * 8 + e : 16 + hsel * 8 + (e - 8))];

  const float scale = 0.044194173824159216f;   // 1/sqrt(512)
  v8f oacc[4] = { {}, {}, {}, {} };
  float mrun = -__builtin_inff();
  float lrun = 0.f;

  for (int j0 = 0; j0 < SEQ; j0 += 32) {
    // ---- transposed score tiles: T[ms] = K rows (j0+ms*16..+15) x Q^T ----
    v8f t0 = {}, t1 = {};
#pragma unroll
    for (int ms = 0; ms < 2; ++ms) {
      v8f acc = {};
#pragma unroll
      for (int dc = 0; dc < 2; ++dc) {
        const bf16_t* kr = kb + (size_t)(j0 + ms * 16 + rlane) * D_MODEL;
        int kofs = dc * 32 + hsel * 8;
        v8bf lo = *(const v8bf*)(kr + kofs);
        v8bf hi = *(const v8bf*)(kr + kofs + 16);
        v16bf afrag;
#pragma unroll
        for (int i = 0; i < 8; ++i) { afrag[i] = lo[i]; afrag[8 + i] = hi[i]; }
        acc = wmma_bf16(afrag, qt[dc], acc);
      }
      if (ms == 0) t0 = acc; else t1 = acc;
    }

    // ---- online softmax: lane holds 16 of this step's 32 scores for q-row
    //      (lane&15); lanes L and L+16 pair up via shfl_xor(16). ----
    float p0[8], p1[8];
    float mc = -__builtin_inff();
#pragma unroll
    for (int r = 0; r < 8; ++r) {
      p0[r] = t0[r] * scale;
      p1[r] = t1[r] * scale;
      mc = fmaxf(mc, fmaxf(p0[r], p1[r]));
    }
    mc = fmaxf(mc, __shfl_xor(mc, 16, 32));
    float mnew = fmaxf(mrun, mc);
    float f = __expf(mrun - mnew);
    float ps = 0.f;
#pragma unroll
    for (int r = 0; r < 8; ++r) {
      p0[r] = __expf(p0[r] - mnew);
      p1[r] = __expf(p1[r] - mnew);
      ps += p0[r] + p1[r];
    }
    ps += __shfl_xor(ps, 16, 32);
    lrun = lrun * f + ps;
    mrun = mnew;

    // Rescale O accumulators: row (r + hsel*8) factor lives at lane (r + hsel*8)
#pragma unroll
    for (int r = 0; r < 8; ++r) {
      float fr = __shfl(f, r + hsel * 8, 32);
      oacc[0][r] = oacc[0][r] * fr;
      oacc[1][r] = oacc[1][r] * fr;
      oacc[2][r] = oacc[2][r] * fr;
      oacc[3][r] = oacc[3][r] * fr;
    }

    // ---- repack P into bf16 A-frag: pure per-lane (layout identity) ----
    v16bf pa;
#pragma unroll
    for (int r = 0; r < 8; ++r) { pa[r] = (bf16_t)p0[r]; pa[8 + r] = (bf16_t)p1[r]; }

    // ---- O += P(16x32) * V(32x64): 4 WMMAs ----
    const bf16_t* vr = vb + (size_t)j0 * D_MODEL + rlane;
#pragma unroll
    for (int nb = 0; nb < 4; ++nb) {
      v16bf bfrag;
#pragma unroll
      for (int s = 0; s < 16; ++s)
        bfrag[s] = vr[(size_t)(s < 8 ? hsel * 8 + s : 16 + hsel * 8 + (s - 8)) * D_MODEL + nb * 16];
      oacc[nb] = wmma_bf16(pa, bfrag, oacc[nb]);
    }
  }

  // ---- finalize: divide by row sums, store bf16 ----
  float inv = 1.0f / lrun;
  bf16_t* ob = o + ((size_t)b * SEQ + qrow0) * D_MODEL + h * HDIM;
#pragma unroll
  for (int r = 0; r < 8; ++r) {
    float lr = __shfl(inv, r + hsel * 8, 32);
    int rr = r + hsel * 8;
#pragma unroll
    for (int nb = 0; nb < 4; ++nb)
      ob[(size_t)rr * D_MODEL + nb * 16 + rlane] = (bf16_t)(oacc[nb][r] * lr);
  }
}

// ---------------------------------------------------------------------------
// Launcher
// ---------------------------------------------------------------------------
extern "C" void kernel_launch(void* const* d_in, const int* in_sizes, int n_in,
                              void* d_out, int out_size, void* d_ws, size_t ws_size,
                              hipStream_t stream) {
  (void)in_sizes; (void)n_in; (void)out_size; (void)ws_size;

  const float* x    = (const float*)d_in[0];
  // d_in[1]=t, d_in[2]=actions : unused by the block
  const float* K_W  = (const float*)d_in[3];
  const float* K_b  = (const float*)d_in[4];
  const float* Q_W  = (const float*)d_in[5];
  const float* Q_b  = (const float*)d_in[6];
  const float* V_W  = (const float*)d_in[7];
  const float* V_b  = (const float*)d_in[8];
  const float* O_W  = (const float*)d_in[9];
  const float* O_b  = (const float*)d_in[10];
  const float* n1w  = (const float*)d_in[11];
  const float* n1b  = (const float*)d_in[12];
  const float* n2w  = (const float*)d_in[13];
  const float* n2b  = (const float*)d_in[14];
  const float* W1   = (const float*)d_in[15];
  const float* b1   = (const float*)d_in[16];
  const float* W2   = (const float*)d_in[17];
  const float* b2   = (const float*)d_in[18];
  float* out = (float*)d_out;

  char* ws = (char*)d_ws;
  const size_t WMAT = (size_t)D_MODEL * D_MODEL * sizeof(bf16_t);   // 512 KB
  const size_t ACT  = (size_t)TOK * D_MODEL * sizeof(bf16_t);       // 16 MB
  bf16_t* Wq   = (bf16_t*)(ws);
  bf16_t* Wk   = (bf16_t*)(ws + 1 * WMAT);
  bf16_t* Wv   = (bf16_t*)(ws + 2 * WMAT);
  bf16_t* Wo   = (bf16_t*)(ws + 3 * WMAT);
  bf16_t* W1t  = (bf16_t*)(ws + 4 * WMAT);
  bf16_t* W2t  = (bf16_t*)(ws + 5 * WMAT);
  char*   base = ws + 6 * WMAT;
  bf16_t* RES  = (bf16_t*)(base);             // LN1 output
  bf16_t* Qa   = (bf16_t*)(base + 1 * ACT);
  bf16_t* Ka   = (bf16_t*)(base + 2 * ACT);
  bf16_t* Va   = (bf16_t*)(base + 3 * ACT);
  bf16_t* OA   = (bf16_t*)(base + 4 * ACT);   // attention output
  bf16_t* Mb   = (bf16_t*)(base + 5 * ACT);   // LN2 output
  bf16_t* H1   = (bf16_t*)(base + 6 * ACT);   // GELU(fc1) output
  float*  ORES = (float*) (base + 7 * ACT);   // o + x residual (fp32, 32 MB)

  // 1. repack weights to bf16 K-major
  prep_weights<<<dim3(1024, 6), 256, 0, stream>>>(Q_W, K_W, V_W, O_W, W1, W2,
                                                  Wq, Wk, Wv, Wo, W1t, W2t);
  // 2. LN1 on transposed x
  ln_kernel<true><<<2048, 256, 0, stream>>>(x, n1w, n1b, RES);
  // 3. Q/K/V projections (WMMA + TDM-staged weights)
  gemm_wmma<0><<<2048, 128, 0, stream>>>(RES, Wq, Q_b, nullptr, Qa, nullptr);
  gemm_wmma<0><<<2048, 128, 0, stream>>>(RES, Wk, K_b, nullptr, Ka, nullptr);
  gemm_wmma<0><<<2048, 128, 0, stream>>>(RES, Wv, V_b, nullptr, Va, nullptr);
  // 4. flash attention (WMMA both matmuls)
  attn_kernel<<<2048, 128, 0, stream>>>(Qa, Ka, Va, OA);
  // 5. output projection + residual with x
  gemm_wmma<1><<<2048, 128, 0, stream>>>(OA, Wo, O_b, x, nullptr, ORES);
  // 6. LN2
  ln_kernel<false><<<2048, 256, 0, stream>>>(ORES, n2w, n2b, Mb);
  // 7. MLP fc1 + exact GELU
  gemm_wmma<2><<<2048, 128, 0, stream>>>(Mb, W1t, b1, nullptr, H1, nullptr);
  // 8. MLP fc2 + residual, transposed store to [B,C,H,W]
  gemm_wmma<3><<<2048, 128, 0, stream>>>(H1, W2t, b2, ORES, nullptr, out);
}
